// Decoder_66443144069668
// MI455X (gfx1250) — matrix operand order repaired and checked by reading, not compiled, not run on hardware
//
#include <hip/hip_runtime.h>
#include <hip/hip_bf16.h>

typedef __attribute__((ext_vector_type(16))) __bf16 v16bf;
typedef __attribute__((ext_vector_type(8)))  float  v8f;

constexpr int S1 = 64, S2 = 1024, NROWS = S1 * S2;   // 65536
constexpr int DD = 50;     // latent dim
constexpr int HH = 80;     // hidden
constexpr int TT = 150;    // topics
constexpr int VV = 2000;   // vocab
constexpr int NTILES = VV / 16;   // 125 N-tiles of 16 cols
constexpr int KCH = 5;            // ceil(150/32) K-chunks of 32
constexpr int BS = 512;           // 16 waves per block
constexpr int NWAVE = BS / 32;    // 16
constexpr int TPW = 8;            // tiles per wave (16*8 = 128 >= 125)
constexpr float BN_EPS = 1e-5f;

__device__ __forceinline__ unsigned short f2bf(float f) {
  unsigned int u = __float_as_uint(f);
  u += 0x7FFFu + ((u >> 16) & 1u);           // round-to-nearest-even
  return (unsigned short)(u >> 16);
}

// ---------------------------------------------------------------------------
// Pack W3 (f32 row-major 150x2000) into WMMA bf16 B-fragment layout:
// B_pack[((kc*125 + nt)*32 + lane)*16 + e] = W3[kc*32 + kbase(lane) + e][nt*16 + (lane&15)]
// kbase = 0 for lanes 0-15, 16 for lanes 16-31 (contiguous k order, same as the A pack).
// Also zeroes the column-sum / sum-of-squares accumulators.
// ---------------------------------------------------------------------------
__global__ void k_pack_w3(const float* __restrict__ W3,
                          unsigned short* __restrict__ Bp,
                          float* __restrict__ sums) {
  int idx = blockIdx.x * 256 + threadIdx.x;
  if (idx < 2 * VV) sums[idx] = 0.0f;           // sums[0..1999], sumsq[2000..3999]
  if (idx >= KCH * NTILES * 32 * 16) return;
  int e    = idx & 15;
  int lane = (idx >> 4) & 31;
  int rest = idx >> 9;
  int nt   = rest % NTILES;
  int kc   = rest / NTILES;
  int n    = nt * 16 + (lane & 15);
  int K    = kc * 32 + ((lane < 16) ? 0 : 16) + e;
  Bp[idx] = (K < TT) ? f2bf(W3[K * VV + n]) : (unsigned short)0;
}

// ---------------------------------------------------------------------------
// out_notes = zV @ zW^T + bu bp^T   (tiny: 65536 dots of length 50)
// ---------------------------------------------------------------------------
__global__ void k_notes(const float* __restrict__ zV, const float* __restrict__ zW,
                        const float* __restrict__ bu, const float* __restrict__ bp,
                        float* __restrict__ out) {
  int idx = blockIdx.x * 256 + threadIdx.x;
  if (idx >= NROWS) return;
  int i = idx >> 10, j = idx & 1023;
  const float* a = zV + i * DD;
  const float* b = zW + j * DD;
  float acc = bu[i] * bp[j];
  #pragma unroll 10
  for (int k = 0; k < DD; ++k) acc += a[k] * b[k];
  out[idx] = acc;
}

// ---------------------------------------------------------------------------
// Compute p = softmax(relu(pairs@W1+b1)@W2+b2) for a 16-row tile (rows
// i*1024 + jb .. jb+15) and pack it bf16 into the WMMA A-fragment layout.
// cV[h] = b1[h] + zV[i]·W1[0:50,h] is precomputed per block (constant over j).
// All barriers internal; A_lds valid for all waves on return. BS=512 threads.
// ---------------------------------------------------------------------------
__device__ __forceinline__ void compute_p_tile(
    int jb,
    const float* __restrict__ zW, const float* __restrict__ W1,
    const float* __restrict__ W2, const float* __restrict__ b2,
    const float* cV, float* zWl, float* th1, float* th, float* red32,
    unsigned short* A_lds, int tid) {
  // stage the 16 zW rows
  for (int o = tid; o < 16 * DD; o += BS) {
    int r = o / DD, k = o % DD;
    zWl[o] = zW[(jb + r) * DD + k];
  }
  __syncthreads();
  // theta1 = relu(cV + zW-part @ W1[50:100])   (16x80)
  for (int o = tid; o < 16 * HH; o += BS) {
    int r = o / HH, h = o % HH;
    float v = cV[h];
    const float* zr = &zWl[r * DD];
    #pragma unroll 10
    for (int k = 0; k < DD; ++k) v += zr[k] * W1[(DD + k) * HH + h];
    th1[o] = fmaxf(v, 0.0f);
  }
  __syncthreads();
  // theta = theta1 @ W2 + b2   (16x150)
  for (int o = tid; o < 16 * TT; o += BS) {
    int r = o / TT, t = o % TT;
    float v = b2[t];
    const float* hr = &th1[r * HH];
    #pragma unroll 8
    for (int h = 0; h < HH; ++h) v += hr[h] * W2[h * TT + t];
    th[r * TT + t] = v;
  }
  __syncthreads();
  // row softmax over 150, 32 threads per row
  int row = tid >> 5, sub = tid & 31;
  float m = -3.4e38f;
  for (int t = sub; t < TT; t += 32) m = fmaxf(m, th[row * TT + t]);
  red32[row * 32 + sub] = m;
  __syncthreads();
  if (sub == 0) {
    float mm = red32[row * 32];
    for (int s = 1; s < 32; ++s) mm = fmaxf(mm, red32[row * 32 + s]);
    red32[row * 32] = mm;
  }
  __syncthreads();
  float rm = red32[row * 32];
  float ss = 0.0f;
  for (int t = sub; t < TT; t += 32) {
    float e = __expf(th[row * TT + t] - rm);
    th[row * TT + t] = e;
    ss += e;
  }
  __syncthreads();
  red32[row * 32 + sub] = ss;
  __syncthreads();
  if (sub == 0) {
    float s = 0.0f;
    for (int q = 0; q < 32; ++q) s += red32[row * 32 + q];
    red32[row * 32] = s;
  }
  __syncthreads();
  float inv = 1.0f / red32[row * 32];
  for (int t = sub; t < TT; t += 32) th[row * TT + t] *= inv;
  __syncthreads();
  // pack A fragments: A_lds[kc][lane][e] with k = kc*32 + 16*(lane>=16) + e (contiguous order, matches B pack)
  for (int o = tid; o < KCH * 32 * 16; o += BS) {
    int e = o & 15, lane = (o >> 4) & 31, kc = o >> 9;
    int rr = lane & 15;
    int kb = (lane < 16) ? 0 : 16;
    int kk = kb + e;
    int K = kc * 32 + kk;
    A_lds[o] = (K < TT) ? f2bf(th[rr * TT + K]) : (unsigned short)0;
  }
  __syncthreads();
}

// ---------------------------------------------------------------------------
// Pass A: column statistics of hv = p @ W3 + b3 (never materialized).
// Each block covers 64 rows (4 sub-tiles of 16), register-accumulates
// per-column sum / sumsq, then one atomicAdd per column per block.
// ---------------------------------------------------------------------------
__global__ void __launch_bounds__(BS) k_passA(
    const float* __restrict__ zV, const float* __restrict__ zW,
    const float* __restrict__ W1, const float* __restrict__ b1,
    const float* __restrict__ W2, const float* __restrict__ b2,
    const float* __restrict__ b3, const unsigned short* __restrict__ Bp,
    float* __restrict__ sums, float* __restrict__ sumsq) {
  __shared__ float zVi[DD];
  __shared__ float cV[HH];
  __shared__ float zWl[16 * DD];
  __shared__ float th1[16 * HH];
  __shared__ float th[16 * TT];
  __shared__ float red32[16 * 32];
  __shared__ __align__(32) unsigned short A_lds[KCH * 32 * 16];

  int tid = threadIdx.x;
  int lane = tid & 31, wave = tid >> 5;
  int R0 = blockIdx.x * 64;
  int i = R0 >> 10, j0 = R0 & 1023;

  for (int o = tid; o < DD; o += BS) zVi[o] = zV[i * DD + o];
  __syncthreads();
  if (tid < HH) {
    float v = b1[tid];
    #pragma unroll 10
    for (int k = 0; k < DD; ++k) v += zVi[k] * W1[k * HH + tid];
    cV[tid] = v;
  }

  float accS[TPW], accQ[TPW];
  #pragma unroll
  for (int t = 0; t < TPW; ++t) { accS[t] = 0.0f; accQ[t] = 0.0f; }

  for (int sb = 0; sb < 4; ++sb) {
    compute_p_tile(j0 + sb * 16, zW, W1, W2, b2, cV, zWl, th1, th, red32, A_lds, tid);
    // opaque 32-bit offset: stops LICM/CSE from hoisting the (sub-block
    // invariant) B-fragment loads out of the sb loop, while keeping Bp in
    // the global address space (global_load, not flat_load).
    unsigned int off0 = 0;
    asm volatile("" : "+v"(off0));
    const unsigned short* Bq = Bp + off0;
    v16bf a[KCH];
    #pragma unroll
    for (int kc = 0; kc < KCH; ++kc)
      a[kc] = *reinterpret_cast<const v16bf*>(&A_lds[(kc * 32 + lane) * 16]);
    #pragma unroll
    for (int t = 0; t < TPW; ++t) {
      int nt = wave + t * NWAVE;
      if (nt < NTILES) {
        v8f acc = {0.f, 0.f, 0.f, 0.f, 0.f, 0.f, 0.f, 0.f};
        #pragma unroll
        for (int kc = 0; kc < KCH; ++kc) {
          v16bf b = *reinterpret_cast<const v16bf*>(&Bq[((kc * NTILES + nt) * 32 + lane) * 16]);
          acc = __builtin_amdgcn_wmma_f32_16x16x32_bf16(false, a[kc], false, b,
                                                        (short)0, acc, false, false);
        }
        int n = nt * 16 + (lane & 15);
        float b3n = b3[n];
        float s = 0.0f, q = 0.0f;
        #pragma unroll
        for (int r = 0; r < 8; ++r) {
          float hv = acc[r] + b3n;
          s += hv; q += hv * hv;
        }
        accS[t] += s; accQ[t] += q;
      }
      __builtin_amdgcn_sched_barrier(0);   // keep one tile's B in flight
    }
    __syncthreads();   // all waves done with A_lds before next sub-tile
  }
  // lane l and l+16 share column n; fold then one atomic per column
  #pragma unroll
  for (int t = 0; t < TPW; ++t) {
    int nt = wave + t * NWAVE;
    float s = accS[t] + __shfl_down(accS[t], 16, 32);
    float q = accQ[t] + __shfl_down(accQ[t], 16, 32);
    if (nt < NTILES && lane < 16) {
      int n = nt * 16 + lane;
      atomicAdd(&sums[n], s);
      atomicAdd(&sumsq[n], q);
    }
  }
}

__global__ void k_finalize(const float* __restrict__ sums, const float* __restrict__ sumsq,
                           float* __restrict__ mean, float* __restrict__ invstd) {
  int n = blockIdx.x * 256 + threadIdx.x;
  if (n >= VV) return;
  const float invN = 1.0f / (float)NROWS;
  float m = sums[n] * invN;
  float var = sumsq[n] * invN - m * m;     // biased variance
  mean[n] = m;
  invstd[n] = rsqrtf(var + BN_EPS);
}

// ---------------------------------------------------------------------------
// Pass B: recompute hv tile (16 rows x 2000) in registers, fuse batchnorm +
// row-softmax, stream out_text with NT stores. D-matrix layout: VGPR r ->
// M = r + 8*(lane>=16), N = lane&15 (ISA 7.12.2). 16 waves/block -> 8 tiles
// (64 acc VGPRs) per lane.
// ---------------------------------------------------------------------------
__global__ void __launch_bounds__(BS) k_passB(
    const float* __restrict__ zV, const float* __restrict__ zW,
    const float* __restrict__ W1, const float* __restrict__ b1,
    const float* __restrict__ W2, const float* __restrict__ b2,
    const float* __restrict__ b3, const unsigned short* __restrict__ Bp,
    const float* __restrict__ mean, const float* __restrict__ invstd,
    float* __restrict__ out_text) {
  __shared__ float zVi[DD];
  __shared__ float cV[HH];
  __shared__ float zWl[16 * DD];
  __shared__ float th1[16 * HH];
  __shared__ float th[16 * TT];
  __shared__ float red32[16 * 32];
  __shared__ __align__(32) unsigned short A_lds[KCH * 32 * 16];
  __shared__ float b3s[VV], mns[VV], invs[VV];
  __shared__ float sred[16 * NWAVE];
  __shared__ float fin[16];

  int tid = threadIdx.x;
  int lane = tid & 31, wave = tid >> 5;
  int hh = (lane < 16) ? 0 : 1;            // half: D rows 0-7 vs 8-15
  int R0 = blockIdx.x * 16;
  int i = R0 >> 10, j0 = R0 & 1023;

  for (int o = tid; o < VV; o += BS) {
    b3s[o] = b3[o]; mns[o] = mean[o]; invs[o] = invstd[o];
  }
  for (int o = tid; o < DD; o += BS) zVi[o] = zV[i * DD + o];
  __syncthreads();
  if (tid < HH) {
    float v = b1[tid];
    #pragma unroll 10
    for (int k = 0; k < DD; ++k) v += zVi[k] * W1[k * HH + tid];
    cV[tid] = v;
  }

  compute_p_tile(j0, zW, W1, W2, b2, cV, zWl, th1, th, red32, A_lds, tid);

  v16bf a[KCH];
  #pragma unroll
  for (int kc = 0; kc < KCH; ++kc)
    a[kc] = *reinterpret_cast<const v16bf*>(&A_lds[(kc * 32 + lane) * 16]);

  v8f acc[TPW];
  #pragma unroll
  for (int t = 0; t < TPW; ++t) {
    int nt = wave + t * NWAVE;
    if (nt < NTILES) {
      v8f c = {0.f, 0.f, 0.f, 0.f, 0.f, 0.f, 0.f, 0.f};
      #pragma unroll
      for (int kc = 0; kc < KCH; ++kc) {
        v16bf b = *reinterpret_cast<const v16bf*>(&Bp[((kc * NTILES + nt) * 32 + lane) * 16]);
        c = __builtin_amdgcn_wmma_f32_16x16x32_bf16(false, a[kc], false, b,
                                                    (short)0, c, false, false);
      }
      acc[t] = c;
    }
    __builtin_amdgcn_sched_barrier(0);     // keep one tile's B in flight
  }
  // batchnorm in registers
  #pragma unroll
  for (int t = 0; t < TPW; ++t) {
    int nt = wave + t * NWAVE;
    if (nt < NTILES) {
      int n = nt * 16 + (lane & 15);
      float off = b3s[n] - mns[n];
      float sc = invs[n];
      #pragma unroll
      for (int r = 0; r < 8; ++r) acc[t][r] = (acc[t][r] + off) * sc;
    }
  }
  // ---- row max across 2000 columns ----
  float rmax[8];
  #pragma unroll
  for (int r = 0; r < 8; ++r) rmax[r] = -3.4e38f;
  #pragma unroll
  for (int t = 0; t < TPW; ++t)
    if (wave + t * NWAVE < NTILES) {
      #pragma unroll
      for (int r = 0; r < 8; ++r) rmax[r] = fmaxf(rmax[r], acc[t][r]);
    }
  #pragma unroll
  for (int m = 1; m < 16; m <<= 1) {
    #pragma unroll
    for (int r = 0; r < 8; ++r) rmax[r] = fmaxf(rmax[r], __shfl_xor(rmax[r], m, 32));
  }
  if ((lane & 15) == 0) {
    #pragma unroll
    for (int r = 0; r < 8; ++r) sred[(hh * 8 + r) * NWAVE + wave] = rmax[r];
  }
  __syncthreads();
  if (tid < 16) {
    float v = sred[tid * NWAVE];
    for (int w = 1; w < NWAVE; ++w) v = fmaxf(v, sred[tid * NWAVE + w]);
    fin[tid] = v;
  }
  __syncthreads();
  float fm[8];
  #pragma unroll
  for (int r = 0; r < 8; ++r) fm[r] = fin[hh * 8 + r];
  // ---- exp and row sum ----
  float esum[8];
  #pragma unroll
  for (int r = 0; r < 8; ++r) esum[r] = 0.0f;
  #pragma unroll
  for (int t = 0; t < TPW; ++t)
    if (wave + t * NWAVE < NTILES) {
      #pragma unroll
      for (int r = 0; r < 8; ++r) {
        float e = __expf(acc[t][r] - fm[r]);
        acc[t][r] = e;
        esum[r] += e;
      }
    }
  #pragma unroll
  for (int m = 1; m < 16; m <<= 1) {
    #pragma unroll
    for (int r = 0; r < 8; ++r) esum[r] += __shfl_xor(esum[r], m, 32);
  }
  __syncthreads();   // sred reuse
  if ((lane & 15) == 0) {
    #pragma unroll
    for (int r = 0; r < 8; ++r) sred[(hh * 8 + r) * NWAVE + wave] = esum[r];
  }
  __syncthreads();
  if (tid < 16) {
    float v = 0.0f;
    for (int w = 0; w < NWAVE; ++w) v += sred[tid * NWAVE + w];
    fin[tid] = v;
  }
  __syncthreads();
  float rs[8];
  #pragma unroll
  for (int r = 0; r < 8; ++r) rs[r] = 1.0f / fin[hh * 8 + r];
  // ---- write out (non-temporal: streaming output, never re-read) ----
  #pragma unroll
  for (int t = 0; t < TPW; ++t) {
    int nt = wave + t * NWAVE;
    if (nt < NTILES) {
      int n = nt * 16 + (lane & 15);
      #pragma unroll
      for (int r = 0; r < 8; ++r) {
        int M = hh * 8 + r;
        __builtin_nontemporal_store(acc[t][r] * rs[r],
                                    &out_text[(size_t)(R0 + M) * VV + n]);
      }
    }
  }
}

// ---------------------------------------------------------------------------
extern "C" void kernel_launch(void* const* d_in, const int* in_sizes, int n_in,
                              void* d_out, int out_size, void* d_ws, size_t ws_size,
                              hipStream_t stream) {
  const float* zV = (const float*)d_in[0];
  const float* zW = (const float*)d_in[1];
  const float* bu = (const float*)d_in[2];
  const float* bp = (const float*)d_in[3];
  const float* W1 = (const float*)d_in[4];
  const float* b1 = (const float*)d_in[5];
  const float* W2 = (const float*)d_in[6];
  const float* b2 = (const float*)d_in[7];
  const float* W3 = (const float*)d_in[8];
  const float* b3 = (const float*)d_in[9];

  float* out = (float*)d_out;
  float* out_notes = out;                 // 65536
  float* out_text  = out + NROWS;         // 65536 x 2000

  // workspace layout
  unsigned short* Bp = (unsigned short*)d_ws;                 // 320000 bf16 = 640000 B
  float* wsf   = (float*)((char*)d_ws + 640000);
  float* sums  = wsf;            // [2000]
  float* sumsq = wsf + VV;       // [2000]
  float* meanv = wsf + 2 * VV;   // [2000]
  float* invsd = wsf + 3 * VV;   // [2000]

  k_pack_w3<<<(KCH * NTILES * 32 * 16 + 255) / 256, 256, 0, stream>>>(W3, Bp, sums);
  k_notes<<<(NROWS + 255) / 256, 256, 0, stream>>>(zV, zW, bu, bp, out_notes);
  k_passA<<<NROWS / 64, BS, 0, stream>>>(zV, zW, W1, b1, W2, b2, b3, Bp, sums, sumsq);
  k_finalize<<<(VV + 255) / 256, 256, 0, stream>>>(sums, sumsq, meanv, invsd);
  k_passB<<<NROWS / 16, BS, 0, stream>>>(zV, zW, W1, b1, W2, b2, b3, Bp, meanv, invsd, out_text);
}
